// SparseMoAE_5506148073583
// MI455X (gfx1250) — matrix operand order repaired and checked by reading, not compiled, not run on hardware
//
#include <hip/hip_runtime.h>
#include <hip/hip_bf16.h>

#define T_TOK 32768
#define DD 1024
#define HH 256
#define EE 8
#define KTOP 2
#define CAP 8192

typedef __attribute__((ext_vector_type(16))) __bf16 v16bf;
typedef __attribute__((ext_vector_type(8)))  __bf16 v8bf;
typedef __attribute__((ext_vector_type(4)))  __bf16 v4bf;
typedef __attribute__((ext_vector_type(8)))  float  v8f;

__device__ __forceinline__ v8f wmma_bf16(v16bf a, v16bf b, v8f c) {
    return __builtin_amdgcn_wmma_f32_16x16x32_bf16(
        /*neg_a=*/false, a, /*neg_b=*/false, b,
        /*c_mod=*/(short)0, c, /*reuse_a=*/false, /*reuse_b=*/false);
}

// ---------------------------------------------------------------------------
// Kernel 0: convert W1/W2 fp32 -> bf16, pre-swizzled into WMMA B-fragment
// order so GEMM lanes can load v16bf fragments straight from global.
//   W1s layout: [e][ks:32][nt:16][lane:32][idx:16]
//   W2s layout: [e][ks: 8][nt:64][lane:32][idx:16]
//   fragment map (B 32x16): lane = nn + 16*(k>>4), idx = k&15
// ---------------------------------------------------------------------------
__global__ __launch_bounds__(256) void convert_w_kernel(
    const float* __restrict__ W1, const float* __restrict__ W2,
    __bf16* __restrict__ W1s, __bf16* __restrict__ W2s, int n)
{
    int i = blockIdx.x * 256 + threadIdx.x;
    if (i >= n) return;
    {   // W1: D x HID per expert
        int idx = i & 15, lane = (i >> 4) & 31, nt = (i >> 9) & 15;
        int ks = (i >> 13) & 31, e = i >> 18;
        int k = idx + 16 * (lane >> 4);
        int d = ks * 32 + k;
        int h = nt * 16 + (lane & 15);
        W1s[i] = (__bf16)W1[((size_t)e * DD + d) * HH + h];
    }
    {   // W2: HID x D per expert
        int idx = i & 15, lane = (i >> 4) & 31, nt = (i >> 9) & 63;
        int ks = (i >> 15) & 7, e = i >> 18;
        int k = idx + 16 * (lane >> 4);
        int hk = ks * 32 + k;
        int dn = nt * 16 + (lane & 15);
        W2s[i] = (__bf16)W2[((size_t)e * HH + hk) * DD + dn];
    }
}

// ---------------------------------------------------------------------------
// Kernel 1: router. One wave per token. W (16 x 1024 combined Wg|Wn) in LDS.
// ---------------------------------------------------------------------------
__global__ __launch_bounds__(256) void router_kernel(
    const float* __restrict__ x, const float* __restrict__ noise,
    const float* __restrict__ Wg, const float* __restrict__ bg,
    const float* __restrict__ Wn, const float* __restrict__ bn,
    int* __restrict__ top_i, float* __restrict__ gate, int* __restrict__ inv_slot)
{
    __shared__ float Wc[16][DD];   // 64 KB: rows 0-7 = Wg cols, 8-15 = Wn cols
    int tid = threadIdx.x;
    for (int i = tid; i < 16 * DD; i += 256) {
        int c = i >> 10, d = i & (DD - 1);
        Wc[c][d] = (c < 8) ? Wg[d * EE + c] : Wn[d * EE + (c - 8)];
    }
    __syncthreads();

    int wv = tid >> 5, lane = tid & 31;
    long t = (long)blockIdx.x * 8 + wv;
    const float* xr = x + t * DD;

    float acc[16];
#pragma unroll
    for (int c = 0; c < 16; c++) acc[c] = 0.f;
    for (int i = 0; i < DD / 32; i++) {
        int d = i * 32 + lane;
        float xv = xr[d];
#pragma unroll
        for (int c = 0; c < 16; c++) acc[c] += xv * Wc[c][d];
    }
#pragma unroll
    for (int c = 0; c < 16; c++) {
#pragma unroll
        for (int off = 16; off > 0; off >>= 1)
            acc[c] += __shfl_down(acc[c], off, 32);
    }
    if (lane == 0) {
        float noisy[8];
#pragma unroll
        for (int e = 0; e < 8; e++) {
            float lg = acc[e] + bg[e];
            float nl = acc[8 + e] + bn[e];
            float sp = (nl > 20.f) ? nl : log1pf(expf(nl));   // softplus
            noisy[e] = lg + noise[t * 8 + e] * sp;
        }
        // top-2 (ties -> lower index, matching lax.top_k)
        int i0 = 0; float v0 = noisy[0];
#pragma unroll
        for (int e = 1; e < 8; e++) if (noisy[e] > v0) { v0 = noisy[e]; i0 = e; }
        int i1 = (i0 == 0) ? 1 : 0; float v1 = noisy[i1];
#pragma unroll
        for (int e = 0; e < 8; e++)
            if (e != i0 && e != i1 && noisy[e] > v1) { v1 = noisy[e]; i1 = e; }
        float g0 = 1.f / (1.f + expf(v1 - v0));
        float g1 = 1.f - g0;
        top_i[2 * t] = i0;  top_i[2 * t + 1] = i1;
        gate[2 * t] = g0;   gate[2 * t + 1] = g1;
        inv_slot[2 * t] = -1; inv_slot[2 * t + 1] = -1;
    }
}

// ---------------------------------------------------------------------------
// Kernel 2: per-expert capacity scan in flat-token order. 1 block / expert.
// ---------------------------------------------------------------------------
__global__ __launch_bounds__(1024) void scan_kernel(
    const int* __restrict__ top_i, const float* __restrict__ gate,
    int* __restrict__ inv_slot, int* __restrict__ dispatch_tok,
    float* __restrict__ slot_gate, int* __restrict__ counts)
{
    int e = blockIdx.x;
    int tid = threadIdx.x, lane = tid & 31, wv = tid >> 5;
    __shared__ int wsum[32];
    __shared__ int carry;
    int base = 0;
    for (int c0 = 0; c0 < T_TOK; c0 += 1024) {
        int t = c0 + tid;
        int i0 = top_i[2 * t], i1 = top_i[2 * t + 1];
        bool flag = (i0 == e) || (i1 == e);
        unsigned mask = (unsigned)__ballot(flag);          // wave32
        int lanepfx = __popc(mask & ((1u << lane) - 1u));
        if (lane == 0) wsum[wv] = __popc(mask);
        __syncthreads();
        if (tid == 0) {
            int s = 0;
#pragma unroll
            for (int i = 0; i < 32; i++) { int v = wsum[i]; wsum[i] = s; s += v; }
            carry = s;
        }
        __syncthreads();
        int rank = base + wsum[wv] + lanepfx;
        if (flag && rank < CAP) {
            int k = (i0 == e) ? 0 : 1;
            int slot = e * CAP + rank;
            dispatch_tok[slot] = t;
            slot_gate[slot]    = gate[2 * t + k];
            inv_slot[2 * t + k] = slot;
        }
        base += carry;
        __syncthreads();
    }
    if (tid == 0) counts[e] = (base < CAP) ? base : CAP;
}

// ---------------------------------------------------------------------------
// Kernel 3: fc1 + GELU. Tile M=64 x N=256, K=1024 stepped by 32.
// A: gathered fp32 rows -> bf16 fragments in LDS (double-buffered, one
//    ds_store_b128 per thread per step). B: v16bf fragments direct from
//    pre-swizzled global (L2-resident).
// ---------------------------------------------------------------------------
__global__ __launch_bounds__(256) void fc1_kernel(
    const float* __restrict__ xf, const __bf16* __restrict__ W1s,
    const float* __restrict__ b1, const int* __restrict__ dispatch_tok,
    const int* __restrict__ counts, __bf16* __restrict__ H)
{
    int e = blockIdx.y;
    int cnt = counts[e];
    int mbase = blockIdx.x * 64;
    if (mbase >= cnt) return;                     // block-uniform exit

    __shared__ alignas(32) __bf16 Alds[2][4][32][16];   // 8 KB double buffer
    __shared__ int toks[64];

    int tid = threadIdx.x;
    if (tid < 64) {
        int slot = mbase + tid;
        toks[tid] = (slot < cnt) ? dispatch_tok[e * CAP + slot] : 0;
    }
    __syncthreads();

    int wv = tid >> 5, lane = tid & 31;
    int mt = wv >> 1, nh = wv & 1;

    // A-staging role: thread owns (row am, k-octet koct)
    int am = tid >> 2;                 // 0..63
    int koct = tid & 3;                // k = koct*8 .. koct*8+7
    const float* arow = xf + (size_t)toks[am] * DD + koct * 8;
    __bf16* adst = &Alds[0][am >> 4][(am & 15) + 16 * (koct >> 1)][8 * (koct & 1)];
    const size_t bufStride = 4 * 32 * 16;

    v8f acc[8];
#pragma unroll
    for (int j = 0; j < 8; j++)
#pragma unroll
        for (int q = 0; q < 8; q++) acc[j][q] = 0.f;

    const __bf16* Bbase = W1s + (size_t)e * (DD / 32) * 16 * 32 * 16;

    auto stageA = [&](int ks, int buf) {
        float4 f0 = *(const float4*)(arow + ks * 32);
        float4 f1 = *(const float4*)(arow + ks * 32 + 4);
        v8bf h;
        h[0] = (__bf16)f0.x; h[1] = (__bf16)f0.y; h[2] = (__bf16)f0.z; h[3] = (__bf16)f0.w;
        h[4] = (__bf16)f1.x; h[5] = (__bf16)f1.y; h[6] = (__bf16)f1.z; h[7] = (__bf16)f1.w;
        *(v8bf*)(adst + (size_t)buf * bufStride) = h;
    };

    stageA(0, 0);
    __syncthreads();
    for (int ks = 0; ks < DD / 32; ks++) {
        int buf = ks & 1;
        if (ks + 1 < DD / 32) stageA(ks + 1, buf ^ 1);
        v16bf af = *(const v16bf*)(&Alds[buf][mt][lane][0]);
#pragma unroll
        for (int nt = 0; nt < 8; nt++) {
            size_t off = (((size_t)ks * 16 + (nh * 8 + nt)) * 32 + lane) * 16;
            v16bf bfr = *(const v16bf*)(Bbase + off);
            acc[nt] = wmma_bf16(af, bfr, acc[nt]);
        }
        __syncthreads();
    }

    // epilogue: +b1, exact GELU, store bf16 (row-major H)
    int mhi = (lane >> 4) * 8;
    int ncol = lane & 15;
#pragma unroll
    for (int nt = 0; nt < 8; nt++) {
        int n = (nh * 8 + nt) * 16 + ncol;
        float bias = b1[e * HH + n];
#pragma unroll
        for (int j = 0; j < 8; j++) {
            int m = mt * 16 + mhi + j;
            float v = acc[nt][j] + bias;
            float g = 0.5f * v * (1.f + erff(v * 0.70710678118654752f));
            H[((size_t)e * CAP + mbase + m) * HH + n] = (__bf16)g;
        }
    }
}

// ---------------------------------------------------------------------------
// Kernel 4: fc2. Tile M=64 x N=1024 (4 chunks of 256), K=256.
// Whole 64x256 A-panel staged once into LDS fragments (32 KB); the 256-WMMA
// compute loop then runs with zero barriers. B direct from swizzled global.
// ---------------------------------------------------------------------------
__global__ __launch_bounds__(256) void fc2_kernel(
    const __bf16* __restrict__ H, const __bf16* __restrict__ W2s,
    const float* __restrict__ b2, const float* __restrict__ slot_gate,
    const int* __restrict__ counts, __bf16* __restrict__ Yw)
{
    int e = blockIdx.y;
    int cnt = counts[e];
    int mbase = blockIdx.x * 64;
    if (mbase >= cnt) return;

    __shared__ alignas(32) __bf16 Alds8[8][4][32][16];  // 32 KB: all 8 K-steps
    __shared__ float sg[64];

    int tid = threadIdx.x;
    if (tid < 64) {
        int slot = mbase + tid;
        sg[tid] = (slot < cnt) ? slot_gate[e * CAP + slot] : 0.f;
    }

    const __bf16* Abase = H + ((size_t)e * CAP + mbase) * HH;
    // stage all of A: 2048 octets, 8 per thread, 16B global load -> 16B LDS store
#pragma unroll
    for (int u = 0; u < 8; u++) {
        int oi = tid + u * 256;
        int am = oi >> 5;                 // 0..63
        int rem = oi & 31;
        int ks = rem >> 2, koct = rem & 3;
        v8bf h = *(const v8bf*)(Abase + (size_t)am * HH + ks * 32 + koct * 8);
        *(v8bf*)(&Alds8[ks][am >> 4][(am & 15) + 16 * (koct >> 1)][8 * (koct & 1)]) = h;
    }
    __syncthreads();

    int wv = tid >> 5, lane = tid & 31;
    int mt = wv >> 1, nh = wv & 1;
    int mhi = (lane >> 4) * 8;
    int ncol = lane & 15;

    const __bf16* Bbase = W2s + (size_t)e * (HH / 32) * 64 * 32 * 16;

    for (int ch = 0; ch < 4; ch++) {
        v8f acc[8];
#pragma unroll
        for (int j = 0; j < 8; j++)
#pragma unroll
            for (int q = 0; q < 8; q++) acc[j][q] = 0.f;

#pragma unroll
        for (int ks = 0; ks < HH / 32; ks++) {
            v16bf af = *(const v16bf*)(&Alds8[ks][mt][lane][0]);
#pragma unroll
            for (int nt2 = 0; nt2 < 8; nt2++) {
                int nt = ch * 16 + nh * 8 + nt2;
                size_t off = (((size_t)ks * 64 + nt) * 32 + lane) * 16;
                v16bf bfr = *(const v16bf*)(Bbase + off);
                acc[nt2] = wmma_bf16(af, bfr, acc[nt2]);
            }
        }
        // epilogue: +b2, scale by gate, store bf16
#pragma unroll
        for (int nt2 = 0; nt2 < 8; nt2++) {
            int gn = ch * 256 + (nh * 8 + nt2) * 16 + ncol;
            float bias = b2[e * DD + gn];
#pragma unroll
            for (int j = 0; j < 8; j++) {
                int m = mt * 16 + mhi + j;
                float y = acc[nt2][j] + bias;
                Yw[((size_t)e * CAP + mbase + m) * DD + gn] = (__bf16)(sg[m] * y);
            }
        }
    }
}

// ---------------------------------------------------------------------------
// Kernel 5: combine.  out[t] = wsum*x[t] + Yw[s0] + Yw[s1]  (gather, no atomics)
// ---------------------------------------------------------------------------
__global__ __launch_bounds__(256) void combine_kernel(
    const float* __restrict__ x, const float* __restrict__ gate,
    const int* __restrict__ inv_slot, const __bf16* __restrict__ Yw,
    float* __restrict__ out)
{
    long t = blockIdx.x;
    int s0 = inv_slot[2 * t], s1 = inv_slot[2 * t + 1];
    float g0 = gate[2 * t], g1 = gate[2 * t + 1];
    float wsum = (s0 >= 0 ? g0 : 0.f) + (s1 >= 0 ? g1 : 0.f);

    int d0 = threadIdx.x * 4;
    const float4 xv = *(const float4*)(x + t * DD + d0);
    float r0 = xv.x * wsum, r1 = xv.y * wsum, r2 = xv.z * wsum, r3 = xv.w * wsum;
    if (s0 >= 0) {
        v4bf yp = *(const v4bf*)(Yw + (size_t)s0 * DD + d0);
        r0 += (float)yp[0]; r1 += (float)yp[1]; r2 += (float)yp[2]; r3 += (float)yp[3];
    }
    if (s1 >= 0) {
        v4bf yp = *(const v4bf*)(Yw + (size_t)s1 * DD + d0);
        r0 += (float)yp[0]; r1 += (float)yp[1]; r2 += (float)yp[2]; r3 += (float)yp[3];
    }
    float4 o; o.x = r0; o.y = r1; o.z = r2; o.w = r3;
    *(float4*)(out + t * DD + d0) = o;
}

// ---------------------------------------------------------------------------
// Launch
// ---------------------------------------------------------------------------
extern "C" void kernel_launch(void* const* d_in, const int* in_sizes, int n_in,
                              void* d_out, int out_size, void* d_ws, size_t ws_size,
                              hipStream_t stream) {
    const float* x     = (const float*)d_in[0];
    const float* noise = (const float*)d_in[1];
    const float* Wg    = (const float*)d_in[2];
    const float* bg    = (const float*)d_in[3];
    const float* Wn    = (const float*)d_in[4];
    const float* bn    = (const float*)d_in[5];
    const float* W1    = (const float*)d_in[6];
    const float* b1    = (const float*)d_in[7];
    const float* W2    = (const float*)d_in[8];
    const float* b2    = (const float*)d_in[9];
    float* out = (float*)d_out;

    // workspace carve-out (256B aligned slices)
    char* wp = (char*)d_ws;
    auto carve = [&](size_t bytes) {
        void* p = (void*)wp;
        wp += (bytes + 255) & ~(size_t)255;
        return p;
    };
    int*    top_i        = (int*)   carve((size_t)T_TOK * 2 * sizeof(int));
    float*  gate         = (float*) carve((size_t)T_TOK * 2 * sizeof(float));
    int*    inv_slot     = (int*)   carve((size_t)T_TOK * 2 * sizeof(int));
    int*    counts       = (int*)   carve(EE * sizeof(int));
    int*    dispatch_tok = (int*)   carve((size_t)EE * CAP * sizeof(int));
    float*  slot_gate    = (float*) carve((size_t)EE * CAP * sizeof(float));
    __bf16* W1s          = (__bf16*)carve((size_t)EE * DD * HH * sizeof(__bf16));
    __bf16* W2s          = (__bf16*)carve((size_t)EE * HH * DD * sizeof(__bf16));
    __bf16* Hbuf         = (__bf16*)carve((size_t)EE * CAP * HH * sizeof(__bf16));
    __bf16* Yw           = (__bf16*)carve((size_t)EE * CAP * DD * sizeof(__bf16));

    const int nW = EE * DD * HH;   // elements per weight tensor (same for W1/W2)
    convert_w_kernel<<<dim3((nW + 255) / 256), dim3(256), 0, stream>>>(
        W1, W2, W1s, W2s, nW);

    router_kernel<<<dim3(T_TOK / 8), dim3(256), 0, stream>>>(
        x, noise, Wg, bg, Wn, bn, top_i, gate, inv_slot);

    scan_kernel<<<dim3(EE), dim3(1024), 0, stream>>>(
        top_i, gate, inv_slot, dispatch_tok, slot_gate, counts);

    fc1_kernel<<<dim3(CAP / 64, EE), dim3(256), 0, stream>>>(
        x, W1s, b1, dispatch_tok, counts, Hbuf);

    fc2_kernel<<<dim3(CAP / 64, EE), dim3(256), 0, stream>>>(
        Hbuf, W2s, b2, slot_gate, counts, Yw);

    combine_kernel<<<dim3(T_TOK), dim3(256), 0, stream>>>(
        x, gate, inv_slot, Yw, out);
}
